// SimpleHybridModel_89876485636289
// MI455X (gfx1250) — matrix operand order repaired and checked by reading, not compiled, not run on hardware
//
#include <hip/hip_runtime.h>
#include <hip/hip_bf16.h>

#define G_NUM    256
#define H_DIM    256
#define D_DIM    128
#define N_NODES  102400
#define MAXG     8

typedef __attribute__((ext_vector_type(16))) __bf16 v16bf;
typedef __attribute__((ext_vector_type(8)))  float  v8f;
typedef __attribute__((ext_vector_type(4)))  int    int4v;

#define HAS_ASYNC_LDS (__has_builtin(__builtin_amdgcn_global_load_async_to_lds_b128) && \
                       __has_builtin(__builtin_amdgcn_s_wait_asynccnt))

#if HAS_ASYNC_LDS
typedef __attribute__((address_space(1))) int4v g_int4;
typedef __attribute__((address_space(3))) int4v l_int4;
#endif

union Frag {
    unsigned int u[8];
    uint4        q[2];
    v16bf        v;
};

// precise RNE pack (used only in the tiny one-shot weight-prep kernel)
__device__ __forceinline__ unsigned int pack_bf16_rne(float f0, float f1) {
    unsigned int a = __float_as_uint(f0);
    unsigned int b = __float_as_uint(f1);
    a = (a + 0x7FFFu + ((a >> 16) & 1u)) >> 16;
    b = (b + 0x7FFFu + ((b >> 16) & 1u)) & 0xFFFF0000u;
    return (a & 0xFFFFu) | b;
}

// fast pack for the hot path: round-half-up + one v_perm_b32 (3 VALU total)
__device__ __forceinline__ unsigned int pack_bf16_fast(float f0, float f1) {
    unsigned int a = __float_as_uint(f0) + 0x8000u;
    unsigned int b = __float_as_uint(f1) + 0x8000u;
    // result bytes (LSB..MSB) = {a.b2, a.b3, b.b2, b.b3}
    return __builtin_amdgcn_perm(b, a, 0x07060302u);
}

// copy one LDS-image (count uints, multiple of 4) from prepacked global to LDS
template<int NTHREADS>
__device__ __forceinline__ void stage_w(unsigned int* sW, const unsigned int* src,
                                        int count, int tid) {
#if HAS_ASYNC_LDS
    for (int i = tid * 4; i < count; i += NTHREADS * 4)
        __builtin_amdgcn_global_load_async_to_lds_b128(
            (g_int4*)(src + i), (l_int4*)(sW + i), 0, 0);
    __builtin_amdgcn_s_wait_asynccnt(0);
#else
    for (int i = tid * 4; i < count; i += NTHREADS * 4) {
        uint4 t = *(const uint4*)(src + i);
        *(uint4*)(sW + i) = t;
    }
#endif
}

__device__ __forceinline__ void load_bfrag(Frag& f, const unsigned int* sW,
                                           int nt, int col, int kpb) {
    const unsigned int* bp = &sW[(nt * 16 + col) * 36 + kpb];
    f.q[0] = *(const uint4*)bp;
    f.q[1] = *(const uint4*)(bp + 8);
}

// ---------------------------------------------------------------------------
// Prep: convert fp32 W[K,N] (optionally pre-scaled) into LDS-image layout:
//   out[q][n][kp]  (q = K/64 quarters, kpair kp in [0,32), row pitch 36 dwords)
// One block per quarter.
// ---------------------------------------------------------------------------
__global__ void __launch_bounds__(256)
pack_w_kernel(const float* __restrict__ W, unsigned int* __restrict__ out,
              int N, float scale)
{
    const int q = blockIdx.x;
    unsigned int* dst = out + (size_t)q * N * 36;
    for (int i = threadIdx.x; i < N * 32; i += 256) {
        int n  = i % N;
        int kp = i / N;
        int k  = q * 64 + kp * 2;
        dst[n * 36 + kp] = pack_bf16_rne(W[(size_t)k * N + n] * scale,
                                         W[(size_t)(k + 1) * N + n] * scale);
    }
}

// ---------------------------------------------------------------------------
// Kernel 1: fused  h = relu(x @ W_enc + b_enc)  +  segment_sum(h, batch)
//   grid.x = N/256, block = 512 (16 waves), wave handles 16 rows x 256 cols.
//   Prepacked bf16 W staged into LDS per 64-K half; per-graph LDS reduction.
// ---------------------------------------------------------------------------
__global__ void __launch_bounds__(512)
enc_segsum_kernel(const float* __restrict__ x, const int* __restrict__ batch,
                  const unsigned int* __restrict__ Wp, const float* __restrict__ bias,
                  float* __restrict__ seg)
{
    __shared__ unsigned int sW[H_DIM * 36];   // [n][kpair], pitch 36 dwords, 36 KB
    __shared__ float sSeg[MAXG * H_DIM];      // per-graph partial sums, 8 KB

    const int tid   = threadIdx.x;
    const int wave  = tid >> 5;
    const int lane  = tid & 31;
    const int col   = lane & 15;
    const int khalf = lane >> 4;

    const int rowTop = blockIdx.x * 256 + wave * 16;
    const int gBase  = batch[blockIdx.x * 256];

    for (int i = tid; i < MAXG * H_DIM; i += 512) sSeg[i] = 0.0f;

    v8f acc[16] = {};
    const float* xr = x + (size_t)(rowTop + col) * D_DIM;

    for (int kh = 0; kh < 2; ++kh) {
        __syncthreads();
        stage_w<512>(sW, Wp + (size_t)kh * H_DIM * 36, H_DIM * 36, tid);
        __syncthreads();

        for (int ks = 0; ks < 64; ks += 32) {
            // A fragment (16x32 bf16): lane holds row (lane&15),
            // K runs [kb, kb+8) and [kb+16, kb+24) with kb = base + khalf*8
            Frag a;
            int kb = kh * 64 + ks + khalf * 8;
            float4 a0 = *(const float4*)(xr + kb);
            float4 a1 = *(const float4*)(xr + kb + 4);
            float4 a2 = *(const float4*)(xr + kb + 16);
            float4 a3 = *(const float4*)(xr + kb + 20);
            a.u[0] = pack_bf16_fast(a0.x, a0.y); a.u[1] = pack_bf16_fast(a0.z, a0.w);
            a.u[2] = pack_bf16_fast(a1.x, a1.y); a.u[3] = pack_bf16_fast(a1.z, a1.w);
            a.u[4] = pack_bf16_fast(a2.x, a2.y); a.u[5] = pack_bf16_fast(a2.z, a2.w);
            a.u[6] = pack_bf16_fast(a3.x, a3.y); a.u[7] = pack_bf16_fast(a3.z, a3.w);

            // double-buffered B fragments (32x16 bf16): lane holds col (lane&15)
            const int kpb = (ks >> 1) + khalf * 4;
            Frag bA, bB;
            load_bfrag(bA, sW, 0, col, kpb);
            #pragma unroll
            for (int nt = 0; nt < 16; ++nt) {
                Frag& cur = (nt & 1) ? bB : bA;
                Frag& nxt = (nt & 1) ? bA : bB;
                if (nt < 15) load_bfrag(nxt, sW, nt + 1, col, kpb);
                acc[nt] = __builtin_amdgcn_wmma_f32_16x16x32_bf16(
                    false, a.v, false, cur.v, (short)0, acc[nt], false, false);
            }
        }
    }

    // epilogue: bias + relu + per-graph reduction
    int g[8];
    #pragma unroll
    for (int r = 0; r < 8; ++r) g[r] = batch[rowTop + khalf * 8 + r];

    #pragma unroll
    for (int nt = 0; nt < 16; ++nt) {
        float bv = bias[nt * 16 + col];
        #pragma unroll
        for (int r = 0; r < 8; ++r) {
            float v = fmaxf(acc[nt][r] + bv, 0.0f);
            unsigned int li = (unsigned int)(g[r] - gBase);
            if (li < MAXG) {
                __hip_atomic_fetch_add(&sSeg[li * H_DIM + nt * 16 + col], v,
                                       __ATOMIC_RELAXED, __HIP_MEMORY_SCOPE_WORKGROUP);
            } else {
                __hip_atomic_fetch_add(&seg[(size_t)g[r] * H_DIM + nt * 16 + col], v,
                                       __ATOMIC_RELAXED, __HIP_MEMORY_SCOPE_AGENT);
            }
        }
    }

    __syncthreads();
    for (int i = tid; i < MAXG * H_DIM; i += 512) {
        float v = sSeg[i];
        int gg = gBase + (i >> 8);
        if (v != 0.0f && gg < G_NUM)
            __hip_atomic_fetch_add(&seg[(size_t)gg * H_DIM + (i & (H_DIM - 1))], v,
                                   __ATOMIC_RELAXED, __HIP_MEMORY_SCOPE_AGENT);
    }
}

// ---------------------------------------------------------------------------
// Small-GEMM layer:  C = act(A @ Wp + bias)
//   One wave per 16-row x 64-col tile: grid = (M/128, N/64), block = 256.
//   Each block stages only its 64-column slice of each K-quarter (9 KB).
// ---------------------------------------------------------------------------
template<bool RELU>
__global__ void __launch_bounds__(256)
mlp_kernel(const float* __restrict__ A, const unsigned int* __restrict__ Wp,
           const float* __restrict__ bias, float* __restrict__ C, int K, int N)
{
    __shared__ unsigned int sW[64 * 36];      // 64-col slice, pitch 36 dwords

    const int tid   = threadIdx.x;
    const int wave  = tid >> 5;
    const int lane  = tid & 31;
    const int col   = lane & 15;
    const int khalf = lane >> 4;

    const int rowTop   = blockIdx.x * 128 + wave * 16;
    const int colBase  = blockIdx.y * 64;
    const float* ar    = A + (size_t)(rowTop + col) * K;

    v8f acc[4] = {};

    for (int kq = 0; kq < K; kq += 64) {
        __syncthreads();
        stage_w<256>(sW, Wp + ((size_t)(kq >> 6) * N + colBase) * 36, 64 * 36, tid);
        __syncthreads();

        for (int ks = 0; ks < 64; ks += 32) {
            Frag a;
            int kb = kq + ks + khalf * 8;
            float4 a0 = *(const float4*)(ar + kb);
            float4 a1 = *(const float4*)(ar + kb + 4);
            float4 a2 = *(const float4*)(ar + kb + 16);
            float4 a3 = *(const float4*)(ar + kb + 20);
            a.u[0] = pack_bf16_fast(a0.x, a0.y); a.u[1] = pack_bf16_fast(a0.z, a0.w);
            a.u[2] = pack_bf16_fast(a1.x, a1.y); a.u[3] = pack_bf16_fast(a1.z, a1.w);
            a.u[4] = pack_bf16_fast(a2.x, a2.y); a.u[5] = pack_bf16_fast(a2.z, a2.w);
            a.u[6] = pack_bf16_fast(a3.x, a3.y); a.u[7] = pack_bf16_fast(a3.z, a3.w);

            const int kpb = (ks >> 1) + khalf * 4;
            Frag bA, bB;
            load_bfrag(bA, sW, 0, col, kpb);
            #pragma unroll
            for (int nt = 0; nt < 4; ++nt) {
                Frag& cur = (nt & 1) ? bB : bA;
                Frag& nxt = (nt & 1) ? bA : bB;
                if (nt < 3) load_bfrag(nxt, sW, nt + 1, col, kpb);
                acc[nt] = __builtin_amdgcn_wmma_f32_16x16x32_bf16(
                    false, a.v, false, cur.v, (short)0, acc[nt], false, false);
            }
        }
    }

    #pragma unroll
    for (int nt = 0; nt < 4; ++nt) {
        float bv = bias[colBase + nt * 16 + col];
        #pragma unroll
        for (int r = 0; r < 8; ++r) {
            float v = acc[nt][r] + bv;
            if (RELU) v = fmaxf(v, 0.0f);
            int row = rowTop + khalf * 8 + r;
            C[(size_t)row * N + colBase + nt * 16 + col] = v;
        }
    }
}

// ---------------------------------------------------------------------------
extern "C" void kernel_launch(void* const* d_in, const int* in_sizes, int n_in,
                              void* d_out, int out_size, void* d_ws, size_t ws_size,
                              hipStream_t stream)
{
    const float* x     = (const float*)d_in[0];
    // d_in[1] = edge_index: unused by the reference
    const int*   batch = (const int*)d_in[2];
    const float* W_enc = (const float*)d_in[3];
    const float* b_enc = (const float*)d_in[4];
    const float* W_v1  = (const float*)d_in[5];
    const float* b_v1  = (const float*)d_in[6];
    const float* W_v2  = (const float*)d_in[7];
    const float* b_v2  = (const float*)d_in[8];
    const float* W_p1  = (const float*)d_in[9];
    const float* b_p1  = (const float*)d_in[10];
    const float* W_p2  = (const float*)d_in[11];
    const float* b_p2  = (const float*)d_in[12];

    float* seg = (float*)d_ws;                 // [256, 256]
    float* t1  = seg + G_NUM * H_DIM;
    float* t2  = t1  + G_NUM * H_DIM;
    float* t3  = t2  + G_NUM * H_DIM;

    unsigned int* pEnc = (unsigned int*)(t3 + G_NUM * H_DIM);
    unsigned int* pV1  = pEnc + 2 * H_DIM * 36;   // W_enc: K=128 -> 2 quarters
    unsigned int* pV2  = pV1  + 4 * H_DIM * 36;   // 256x256 -> 4 quarters
    unsigned int* pP1  = pV2  + 4 * H_DIM * 36;
    unsigned int* pP2  = pP1  + 4 * H_DIM * 36;   // W_p2: 256x128 -> 4 quarters of N=128

    (void)hipMemsetAsync(seg, 0, (size_t)G_NUM * H_DIM * sizeof(float), stream);

    // one-shot weight packing (1/8 virtual-node scale folded into W_v1: exact in bf16)
    pack_w_kernel<<<2, 256, 0, stream>>>(W_enc, pEnc, H_DIM, 1.0f);
    pack_w_kernel<<<4, 256, 0, stream>>>(W_v1,  pV1,  H_DIM, 0.125f);
    pack_w_kernel<<<4, 256, 0, stream>>>(W_v2,  pV2,  H_DIM, 1.0f);
    pack_w_kernel<<<4, 256, 0, stream>>>(W_p1,  pP1,  H_DIM, 1.0f);
    pack_w_kernel<<<4, 256, 0, stream>>>(W_p2,  pP2,  128,   1.0f);

    enc_segsum_kernel<<<N_NODES / 256, 512, 0, stream>>>(x, batch, pEnc, b_enc, seg);

    // gf = MLP(seg / 8): all 8 virtual nodes identical, mean over V is a no-op
    mlp_kernel<true ><<<dim3(2, 4), 256, 0, stream>>>(seg, pV1, b_v1, t1, H_DIM, H_DIM);
    mlp_kernel<false><<<dim3(2, 4), 256, 0, stream>>>(t1,  pV2, b_v2, t2, H_DIM, H_DIM);
    mlp_kernel<true ><<<dim3(2, 4), 256, 0, stream>>>(t2,  pP1, b_p1, t3, H_DIM, H_DIM);
    mlp_kernel<false><<<dim3(2, 2), 256, 0, stream>>>(t3,  pP2, b_p2, (float*)d_out, H_DIM, 128);
}
